// CompactS6Layer_14121852469706
// MI455X (gfx1250) — matrix operand order repaired and checked by reading, not compiled, not run on hardware
//
#include <hip/hip_runtime.h>

typedef __attribute__((ext_vector_type(16))) _Float16 v16h;
typedef __attribute__((ext_vector_type(8)))  _Float16 v8h;
typedef __attribute__((ext_vector_type(8)))  float    v8f;

#define D_MODEL 1024
#define D_STATE 8
#define BATCH   8
#define SEQ     2048
#define MROWS   (BATCH * SEQ)          // 16384
#define ELEMS   ((size_t)MROWS * D_MODEL)

// ---------------------------------------------------------------------------
// Weight transpose + f32 -> f16 convert:  Wt[n*K + k] = (f16) W[k*N + n]
// ---------------------------------------------------------------------------
__global__ __launch_bounds__(256)
void transpose_cvt_kernel(const float* __restrict__ W, _Float16* __restrict__ Wt,
                          int K, int N, size_t total) {
    size_t i = (size_t)blockIdx.x * 256 + threadIdx.x;
    if (i >= total) return;
    size_t n = i / (size_t)K;
    size_t k = i - n * (size_t)K;
    Wt[i] = (_Float16)W[k * (size_t)N + n];
}

// ---------------------------------------------------------------------------
// LayerNorm over D=1024, one row (b,l) per 256-thread block -> f16 output
// ---------------------------------------------------------------------------
__global__ __launch_bounds__(256)
void layernorm_kernel(const float* __restrict__ x, const float* __restrict__ gamma,
                      const float* __restrict__ beta, _Float16* __restrict__ xh) {
    __shared__ float s1[256];
    __shared__ float s2[256];
    size_t row = blockIdx.x;
    const float* xr = x + row * D_MODEL;
    int t = threadIdx.x;
    float v[4];
    float sum = 0.f, sq = 0.f;
#pragma unroll
    for (int i = 0; i < 4; ++i) {
        v[i] = xr[t + i * 256];
        sum += v[i];
        sq  += v[i] * v[i];
    }
    s1[t] = sum; s2[t] = sq;
    __syncthreads();
    for (int s = 128; s > 0; s >>= 1) {
        if (t < s) { s1[t] += s1[t + s]; s2[t] += s2[t + s]; }
        __syncthreads();
    }
    float mu  = s1[0] * (1.f / D_MODEL);
    float var = s2[0] * (1.f / D_MODEL) - mu * mu;
    float rstd = rsqrtf(var + 1e-5f);
#pragma unroll
    for (int i = 0; i < 4; ++i) {
        int c = t + i * 256;
        xh[row * D_MODEL + c] = (_Float16)((v[i] - mu) * rstd * gamma[c] + beta[c]);
    }
}

// ---------------------------------------------------------------------------
// WMMA GEMM:  C[M,Nw] (f32) = A[M,1024](f16) x Wt[Nw,1024](f16, row=output col)
// Block tile 128x128, 8 waves of 32x64, LDS staged, K-step 64 (2 WMMA chunks
// per stage -> 16 barrier stages). Epilogue uses one per-lane base pointer +
// compile-time immediate offsets; all control flow wave-uniform.
// MODE 0: split N=2048 -> x_in (f32 + f16 copies) and z (f16)
// MODE 1: softplus(v + bias[col]) -> f32  (branchless)
// MODE 2: v + resid[row,col] -> f32 (final output)
// ---------------------------------------------------------------------------
__device__ inline v16h frag_a(const _Float16* base) {
    // ISA 16-bit A layout: elems 0..7 = K kb..kb+7, elems 8..15 = K 16+kb..
    v8h lo = *(const v8h*)(base);
    v8h hi = *(const v8h*)(base + 16);
    v16h r;
#pragma unroll
    for (int i = 0; i < 8; ++i) { r[i] = lo[i]; r[8 + i] = hi[i]; }
    return r;
}
__device__ inline v16h frag_b(const _Float16* base) {
    // ISA 16-bit B layout: elems 0..7 = K kb..kb+7, elems 8..15 = K 16+kb..
    v8h lo = *(const v8h*)(base);
    v8h hi = *(const v8h*)(base + 16);
    v16h r;
#pragma unroll
    for (int i = 0; i < 8; ++i) { r[i] = lo[i]; r[8 + i] = hi[i]; }
    return r;
}

template <int MODE>
__global__ __launch_bounds__(256)
void gemm_wmma_kernel(const _Float16* __restrict__ A, const _Float16* __restrict__ Wt,
                      float* __restrict__ out0, _Float16* __restrict__ out1,
                      _Float16* __restrict__ out2, const float* __restrict__ bias,
                      const float* __restrict__ resid) {
    constexpr int K   = 1024;
    constexpr int KS  = 64;                 // K per LDS stage
    constexpr int LDA = 72;                 // 64 + 8 halves pad (144B rows)
    __shared__ _Float16 lA[128 * LDA];
    __shared__ _Float16 lB[128 * LDA];

    const int tid  = threadIdx.x;
    const int lane = tid & 31;
    const int w    = tid >> 5;
    const int wm   = w & 3;                 // 4 waves along M
    const int wn   = w >> 2;                // 2 waves along N
    const int bm   = blockIdx.y * 128;
    const int bn   = blockIdx.x * 128;

    v8f acc[2][4] = {};

    // cooperative tile loaders: 2 threads / row, 32 halves (64B) per thread per tile
    const int ldRow = tid >> 1;
    const int ldCol = (tid & 1) * 32;
    const _Float16* gA = A  + (size_t)(bm + ldRow) * K + ldCol;
    const _Float16* gB = Wt + (size_t)(bn + ldRow) * K + ldCol;
    _Float16* sA = &lA[ldRow * LDA + ldCol];
    _Float16* sB = &lB[ldRow * LDA + ldCol];

    const int fr = lane & 15;
    const int kb = (lane >> 4) * 8;         // A-frag K base within chunk
    const int kc = (lane >> 4) * 16;        // unused; B frags use kb as well

    for (int k0 = 0; k0 < K; k0 += KS) {
        v8h a0 = *(const v8h*)(gA + k0);
        v8h a1 = *(const v8h*)(gA + k0 + 8);
        v8h a2 = *(const v8h*)(gA + k0 + 16);
        v8h a3 = *(const v8h*)(gA + k0 + 24);
        v8h b0 = *(const v8h*)(gB + k0);
        v8h b1 = *(const v8h*)(gB + k0 + 8);
        v8h b2 = *(const v8h*)(gB + k0 + 16);
        v8h b3 = *(const v8h*)(gB + k0 + 24);
        __syncthreads();
        *(v8h*)(sA)      = a0; *(v8h*)(sA + 8)  = a1;
        *(v8h*)(sA + 16) = a2; *(v8h*)(sA + 24) = a3;
        *(v8h*)(sB)      = b0; *(v8h*)(sB + 8)  = b1;
        *(v8h*)(sB + 16) = b2; *(v8h*)(sB + 24) = b3;
        __syncthreads();

#pragma unroll
        for (int c = 0; c < 2; ++c) {           // two K=32 chunks per stage
            v16h af[2], bf[4];
#pragma unroll
            for (int mi = 0; mi < 2; ++mi)
                af[mi] = frag_a(&lA[(wm * 32 + mi * 16 + fr) * LDA + c * 32 + kb]);
#pragma unroll
            for (int ni = 0; ni < 4; ++ni)
                bf[ni] = frag_b(&lB[(wn * 64 + ni * 16 + fr) * LDA + c * 32 + kb]);
#pragma unroll
            for (int mi = 0; mi < 2; ++mi)
#pragma unroll
                for (int ni = 0; ni < 4; ++ni)
                    acc[mi][ni] = __builtin_amdgcn_wmma_f32_16x16x32_f16(
                        false, af[mi], false, bf[ni], (short)0, acc[mi][ni],
                        false, false);
        }
    }

    // ---- Epilogue ----------------------------------------------------------
    // acc elem r -> M = rowB + mi*16 + r,  N = colB + ni*16  (per-lane bases)
    const int rowB = bm + wm * 32 + ((lane >> 4) * 8);
    const int colB = bn + wn * 64 + (lane & 15);
    const size_t base = (size_t)rowB * D_MODEL + colB;   // valid when colB<1024

    if constexpr (MODE == 0) {
        const bool isXin = (bn + wn * 64) < D_MODEL;     // wave-uniform
        if (isXin) {
            float*    p0 = out0 + base;
            _Float16* p1 = out1 + base;
#pragma unroll
            for (int mi = 0; mi < 2; ++mi)
#pragma unroll
                for (int ni = 0; ni < 4; ++ni)
#pragma unroll
                    for (int r = 0; r < 8; ++r) {
                        const int off = mi * 16 * D_MODEL + r * D_MODEL + ni * 16;
                        float v = acc[mi][ni][r];
                        p0[off] = v;
                        p1[off] = (_Float16)v;
                    }
        } else {
            _Float16* p2 = out2 + ((size_t)rowB * D_MODEL + (colB - D_MODEL));
#pragma unroll
            for (int mi = 0; mi < 2; ++mi)
#pragma unroll
                for (int ni = 0; ni < 4; ++ni)
#pragma unroll
                    for (int r = 0; r < 8; ++r)
                        p2[mi * 16 * D_MODEL + r * D_MODEL + ni * 16] =
                            (_Float16)acc[mi][ni][r];
        }
    } else if constexpr (MODE == 1) {
        float* p0 = out0 + base;
        float bv[4];
#pragma unroll
        for (int ni = 0; ni < 4; ++ni) bv[ni] = bias[colB + ni * 16];
#pragma unroll
        for (int mi = 0; mi < 2; ++mi)
#pragma unroll
            for (int ni = 0; ni < 4; ++ni)
#pragma unroll
                for (int r = 0; r < 8; ++r) {
                    float t = acc[mi][ni][r] + bv[ni];
                    // branchless softplus: max(t,0) + log(1+exp(-|t|))
                    float sp = fmaxf(t, 0.f) + __logf(1.f + __expf(-fabsf(t)));
                    p0[mi * 16 * D_MODEL + r * D_MODEL + ni * 16] = sp;
                }
    } else {
        float* p0 = out0 + base;
        const float* pr = resid + base;
#pragma unroll
        for (int mi = 0; mi < 2; ++mi)
#pragma unroll
            for (int ni = 0; ni < 4; ++ni)
#pragma unroll
                for (int r = 0; r < 8; ++r) {
                    const int off = mi * 16 * D_MODEL + r * D_MODEL + ni * 16;
                    p0[off] = acc[mi][ni][r] + pr[off];
                }
    }
}

// ---------------------------------------------------------------------------
// B_t / C_t projections: wave per row, W_B/W_C staged in LDS, shfl reduce
// ---------------------------------------------------------------------------
__global__ __launch_bounds__(256)
void bc_kernel(const float* __restrict__ xin, const float* __restrict__ WB,
               const float* __restrict__ WC, float* __restrict__ Bt,
               float* __restrict__ Ct) {
    __shared__ float sWB[D_MODEL * D_STATE];
    __shared__ float sWC[D_MODEL * D_STATE];
    for (int i = threadIdx.x; i < D_MODEL * D_STATE; i += 256) {
        sWB[i] = WB[i];
        sWC[i] = WC[i];
    }
    __syncthreads();
    const int w = threadIdx.x >> 5, lane = threadIdx.x & 31;
    size_t row = (size_t)blockIdx.x * 8 + w;
    const float* xr = xin + row * D_MODEL;
    float aB[D_STATE] = {}, aC[D_STATE] = {};
    for (int k = lane; k < D_MODEL; k += 32) {
        float xv = xr[k];
#pragma unroll
        for (int n = 0; n < D_STATE; ++n) {
            aB[n] += xv * sWB[k * D_STATE + n];
            aC[n] += xv * sWC[k * D_STATE + n];
        }
    }
#pragma unroll
    for (int n = 0; n < D_STATE; ++n) {
        for (int off = 16; off > 0; off >>= 1) {
            aB[n] += __shfl_xor(aB[n], off, 32);
            aC[n] += __shfl_xor(aC[n], off, 32);
        }
    }
    if (lane == 0) {
#pragma unroll
        for (int n = 0; n < D_STATE; ++n) {
            Bt[row * D_STATE + n] = aB[n];
            Ct[row * D_STATE + n] = aC[n];
        }
    }
}

// ---------------------------------------------------------------------------
// Selective scan: one lane per (b, d, n) chain; y reduced over the 8-lane
// n-group via shfl_xor. 65536 lanes = 2048 wave32s.
// ---------------------------------------------------------------------------
__global__ __launch_bounds__(256)
void scan_kernel(const float* __restrict__ dt, const float* __restrict__ xin,
                 const float* __restrict__ Bt, const float* __restrict__ Ct,
                 const float* __restrict__ A_log, const float* __restrict__ h_prev,
                 float* __restrict__ y, float* __restrict__ h_out) {
    int t = blockIdx.x * 256 + threadIdx.x;        // 0 .. 65535
    int b = t >> 13;
    int d = (t >> 3) & (D_MODEL - 1);
    int n = t & (D_STATE - 1);

    float A = -__expf(A_log[d * D_STATE + n]);
    float h = h_prev[(size_t)b * D_MODEL * D_STATE + d * D_STATE + n];

    const float* dtp = dt  + (size_t)b * SEQ * D_MODEL + d;
    const float* xip = xin + (size_t)b * SEQ * D_MODEL + d;
    const float* bp  = Bt  + (size_t)b * SEQ * D_STATE + n;
    const float* cp  = Ct  + (size_t)b * SEQ * D_STATE + n;
    float* yp        = y   + (size_t)b * SEQ * D_MODEL + d;

    for (int l = 0; l < SEQ; ++l) {
        if (n == 0 && l + 8 < SEQ) {
            __builtin_prefetch(dtp + ((size_t)(l + 8) << 10), 0, 1);
            __builtin_prefetch(xip + ((size_t)(l + 8) << 10), 0, 1);
        }
        float dtv = dtp[(size_t)l << 10];
        float xv  = xip[(size_t)l << 10];
        float Bv  = bp[l * D_STATE];
        float Cv  = cp[l * D_STATE];
        h = __expf(dtv * A) * h + (dtv * xv) * Bv;
        float yv = h * Cv;
        yv += __shfl_xor(yv, 1, 32);
        yv += __shfl_xor(yv, 2, 32);
        yv += __shfl_xor(yv, 4, 32);
        if (n == 0) yp[(size_t)l << 10] = yv;
    }
    h_out[(size_t)b * D_MODEL * D_STATE + d * D_STATE + n] = h;
}

// ---------------------------------------------------------------------------
// Gated output:  ssm = y * silu(z) + x_in * D_param   -> f16 (for final GEMM)
// ---------------------------------------------------------------------------
__global__ __launch_bounds__(256)
void gate_kernel(const float* __restrict__ y, const _Float16* __restrict__ z_h,
                 const float* __restrict__ xin, const float* __restrict__ Dp,
                 _Float16* __restrict__ ssm_h) {
    size_t i = (size_t)blockIdx.x * 256 + threadIdx.x;
    if (i >= ELEMS) return;
    int c = (int)(i & (D_MODEL - 1));
    float zv  = (float)z_h[i];
    float sil = zv / (1.f + __expf(-zv));
    ssm_h[i] = (_Float16)(y[i] * sil + xin[i] * Dp[c]);
}

// ---------------------------------------------------------------------------
// Host-side launch
// ---------------------------------------------------------------------------
extern "C" void kernel_launch(void* const* d_in, const int* in_sizes, int n_in,
                              void* d_out, int out_size, void* d_ws, size_t ws_size,
                              hipStream_t stream) {
    (void)in_sizes; (void)n_in; (void)out_size; (void)ws_size;

    const float* x         = (const float*)d_in[0];
    const float* h_prev    = (const float*)d_in[1];
    const float* gamma     = (const float*)d_in[2];
    const float* beta      = (const float*)d_in[3];
    const float* W_in_gate = (const float*)d_in[4];
    const float* W_dt      = (const float*)d_in[5];
    const float* b_dt      = (const float*)d_in[6];
    const float* A_log     = (const float*)d_in[7];
    const float* W_B       = (const float*)d_in[8];
    const float* W_C       = (const float*)d_in[9];
    const float* D_param   = (const float*)d_in[10];
    const float* W_out     = (const float*)d_in[11];

    float* out      = (float*)d_out;              // [B*L*D] x_out
    float* h_final  = out + ELEMS;                // [B*D*N]

    // workspace bump allocator (256B aligned regions), total ~330 MB
    size_t off = 0;
    auto take = [&](size_t bytes) -> char* {
        char* r = (char*)d_ws + off;
        off += (bytes + 255) & ~(size_t)255;
        return r;
    };
    _Float16* Wg_t    = (_Float16*)take((size_t)2048 * 1024 * 2);
    _Float16* Wdt_t   = (_Float16*)take((size_t)1024 * 1024 * 2);
    _Float16* Wout_t  = (_Float16*)take((size_t)1024 * 1024 * 2);
    _Float16* xnorm_h = (_Float16*)take(ELEMS * 2);
    float*    xin_f   = (float*)   take(ELEMS * 4);
    _Float16* xin_h   = (_Float16*)take(ELEMS * 2);
    _Float16* z_h     = (_Float16*)take(ELEMS * 2);
    float*    dt_f    = (float*)   take(ELEMS * 4);
    float*    Bt      = (float*)   take((size_t)MROWS * D_STATE * 4);
    float*    Ct      = (float*)   take((size_t)MROWS * D_STATE * 4);
    float*    y_f     = (float*)   take(ELEMS * 4);
    _Float16* ssm_h   = (_Float16*)take(ELEMS * 2);

    // 1) weights -> f16, transposed to [N, K]
    transpose_cvt_kernel<<<(2048 * 1024 + 255) / 256, 256, 0, stream>>>(
        W_in_gate, Wg_t, 1024, 2048, (size_t)2048 * 1024);
    transpose_cvt_kernel<<<(1024 * 1024 + 255) / 256, 256, 0, stream>>>(
        W_dt, Wdt_t, 1024, 1024, (size_t)1024 * 1024);
    transpose_cvt_kernel<<<(1024 * 1024 + 255) / 256, 256, 0, stream>>>(
        W_out, Wout_t, 1024, 1024, (size_t)1024 * 1024);

    // 2) LayerNorm
    layernorm_kernel<<<MROWS, 256, 0, stream>>>(x, gamma, beta, xnorm_h);

    // 3) x_proj = x_norm @ W_in_gate  (N = 2048), split into x_in / z
    dim3 g0(2048 / 128, MROWS / 128);
    gemm_wmma_kernel<0><<<g0, 256, 0, stream>>>(
        xnorm_h, Wg_t, xin_f, xin_h, z_h, nullptr, nullptr);

    // 4) dt = softplus(x_in @ W_dt + b_dt)
    dim3 g1(1024 / 128, MROWS / 128);
    gemm_wmma_kernel<1><<<g1, 256, 0, stream>>>(
        xin_h, Wdt_t, dt_f, nullptr, nullptr, b_dt, nullptr);

    // 5) B_t, C_t
    bc_kernel<<<MROWS / 8, 256, 0, stream>>>(xin_f, W_B, W_C, Bt, Ct);

    // 6) selective scan (writes y and h_final)
    scan_kernel<<<(BATCH * D_MODEL * D_STATE) / 256, 256, 0, stream>>>(
        dt_f, xin_f, Bt, Ct, A_log, h_prev, y_f, h_final);

    // 7) gate: ssm = y * silu(z) + x_in * D
    gate_kernel<<<(unsigned)((ELEMS + 255) / 256), 256, 0, stream>>>(
        y_f, z_h, xin_f, D_param, ssm_h);

    // 8) x_out = x + ssm @ W_out
    gemm_wmma_kernel<2><<<g1, 256, 0, stream>>>(
        ssm_h, Wout_t, out, nullptr, nullptr, nullptr, x);
}
